// MultiModalAttention_23527830847511
// MI455X (gfx1250) — matrix operand order repaired and checked by reading, not compiled, not run on hardware
//
#include <hip/hip_runtime.h>

typedef __bf16 v16bf __attribute__((ext_vector_type(16)));
typedef float  v8f   __attribute__((ext_vector_type(8)));
typedef int    int4v __attribute__((ext_vector_type(4)));

static constexpr int T = 4096;
static constexpr int L = 4096;

#if defined(__has_builtin)
#  if __has_builtin(__builtin_amdgcn_global_load_async_to_lds_b128)
#    define HAVE_ASYNC_LDS 1
#  endif
#  if __has_builtin(__builtin_amdgcn_s_wait_asynccnt)
#    define HAVE_WAIT_ASYNC 1
#  endif
#endif
#ifndef HAVE_ASYNC_LDS
#  define HAVE_ASYNC_LDS 0
#endif
#ifndef HAVE_WAIT_ASYNC
#  define HAVE_WAIT_ASYNC 0
#endif

#if HAVE_ASYNC_LDS
typedef __attribute__((address_space(3))) void*  as3_void;
typedef __attribute__((address_space(3))) int4v* as3_int4;
#endif

// 16-bit A-matrix 16x32 lane layout (contiguous k order, matching B):
// lanes 0-15: element e -> K=e ; lanes 16-31: K=16+e
__device__ __forceinline__ int a_kidx(int e, int h) {
    return 16 * h + e;
}

// ---------------------------------------------------------------------------
// Kernel 0a: fold Wq,bq,Wk,bk into alpha/beta generators.
// P layout per modality (stride 160 floats): ua[64], ub[64], ca, cb
// ---------------------------------------------------------------------------
__global__ void prep_kernel(const float* __restrict__ Wq, const float* __restrict__ bq,
                            const float* __restrict__ Wk1, const float* __restrict__ bk1,
                            const float* __restrict__ Wk2, const float* __restrict__ bk2,
                            float* __restrict__ P) {
    int tid = threadIdx.x;
    if (tid >= 128) return;
    int mod = tid >> 6;
    int d   = tid & 63;
    const float* w  = mod ? Wk2 : Wk1;
    const float* bk = mod ? bk2 : bk1;
    const float* row = Wq + (size_t)d * 128;
    float ua = row[0], ub = 0.f;
    for (int c = 1; c <= 127; ++c) { ua -= row[c] * bk[c - 1]; ub += row[c] * w[c - 1]; }
    P[mod * 160 + d]      = ua;
    P[mod * 160 + 64 + d] = ub;
    if (d == 0) {
        float ca = bq[0], cb = 0.f;
        for (int c = 1; c <= 127; ++c) { ca -= bq[c] * bk[c - 1]; cb += bq[c] * w[c - 1]; }
        ca -= bk[127]; cb += w[127];
        P[mod * 160 + 128] = ca;
        P[mod * 160 + 129] = cb;
    }
}

// ---------------------------------------------------------------------------
// Kernel 0b: pre-swizzle Wv into bf16 WMMA B-fragment layout.
// fragment id f = ((kc/32)*8 + j)*32 + (h*16+m); 16 bf16 per fragment:
//   elem e = Wv[kc + h*16 + e][j*16 + m]
// ---------------------------------------------------------------------------
__global__ void wvprep_kernel(const float* __restrict__ Wv1, const float* __restrict__ Wv2,
                              __bf16* __restrict__ WvB1, __bf16* __restrict__ WvB2) {
    int mod = blockIdx.y;
    const float* Wv = mod ? Wv2 : Wv1;
    __bf16* WvB = mod ? WvB2 : WvB1;
    int dIn = mod ? 64 : 128;
    int nfrag = (dIn / 32) * 8 * 32;
    int f = blockIdx.x * blockDim.x + threadIdx.x;
    if (f >= nfrag) return;
    int lanei = f & 31;
    int j  = (f >> 5) & 7;
    int kc = (f >> 8) * 32;
    int m = lanei & 15, h = lanei >> 4;
    __bf16* outp = WvB + (size_t)f * 16;
    #pragma unroll
    for (int e = 0; e < 16; ++e)
        outp[e] = (__bf16)Wv[(size_t)(kc + h * 16 + e) * 128 + j * 16 + m];
}

// ---------------------------------------------------------------------------
// Kernel 1: per-query alpha/beta + attention window [lo, cend) and hi.
// ---------------------------------------------------------------------------
__global__ void qparam_kernel(const float* __restrict__ ref_data,
                              const float* __restrict__ ref_time,
                              const int*   __restrict__ ref_idx,
                              const float* __restrict__ m1_time, const int* __restrict__ m1_idx,
                              const float* __restrict__ m2_time, const int* __restrict__ m2_idx,
                              const float* __restrict__ P,
                              float* __restrict__ qab, int* __restrict__ qrg) {
    int t = blockIdx.x * blockDim.x + threadIdx.x;
    if (t >= T) return;
    const float* r = ref_data + (size_t)t * 64;
    float qt = ref_time[t];
    int   qid = ref_idx[t];
    for (int mod = 0; mod < 2; ++mod) {
        const float* p = P + mod * 160;
        float a = p[128], b = p[129];
        #pragma unroll 8
        for (int d = 0; d < 64; ++d) { a += r[d] * p[d]; b += r[d] * p[64 + d]; }
        qab[mod * 2 * T + t]     = a;
        qab[mod * 2 * T + T + t] = b;
        const int*   kidx  = mod ? m2_idx  : m1_idx;
        const float* ktime = mod ? m2_time : m1_time;
        int lo, hi;
        { int a0 = 0, b0 = L; while (a0 < b0) { int m = (a0 + b0) >> 1; if (kidx[m] <  qid) a0 = m + 1; else b0 = m; } lo = a0; }
        { int a0 = lo, b0 = L; while (a0 < b0) { int m = (a0 + b0) >> 1; if (kidx[m] <= qid) a0 = m + 1; else b0 = m; } hi = a0; }
        if (lo == hi) { lo = 0; hi = L; }   // fully-masked "safe" path
        int ce;
        { int a0 = lo, b0 = hi; while (a0 < b0) { int m = (a0 + b0) >> 1; if (ktime[m] <= qt) a0 = m + 1; else b0 = m; } ce = a0; }
        qrg[(mod * 3 + 0) * T + t] = lo;
        qrg[(mod * 3 + 1) * T + t] = ce;
        qrg[(mod * 3 + 2) * T + t] = hi;
    }
}

// ---------------------------------------------------------------------------
// Kernel 2: V projection with WMMA bf16 -> column-major bf16 VbT[128][L].
// B fragments come pre-swizzled from WvB (one 32B load per tile per lane).
// ---------------------------------------------------------------------------
__global__ void vproj_kernel(const float* __restrict__ m1_data, const __bf16* __restrict__ WvB1,
                             const float* __restrict__ bv1,
                             const float* __restrict__ m2_data, const __bf16* __restrict__ WvB2,
                             const float* __restrict__ bv2,
                             __bf16* __restrict__ VbT1, __bf16* __restrict__ VbT2) {
    int mod = blockIdx.y;
    const float* data = mod ? m2_data : m1_data;
    const __bf16* WvB = mod ? WvB2 : WvB1;
    const float* bv   = mod ? bv2 : bv1;
    __bf16* VbT = mod ? VbT2 : VbT1;
    int stride = mod ? 66 : 130;
    int dIn    = mod ? 64 : 128;

    int l0 = blockIdx.x * 16;
    int lane = threadIdx.x;
    int m = lane & 15, h = lane >> 4;

    v8f acc[8] = {};
    for (int kc = 0; kc < dIn; kc += 32) {
        v16bf A;
        const float* row = data + (size_t)(l0 + m) * stride + kc;
        #pragma unroll
        for (int e = 0; e < 16; ++e) A[e] = (__bf16)row[a_kidx(e, h)];
        const v16bf* bfrag = (const v16bf*)(WvB + ((size_t)(kc >> 5) * 8 * 32 + lane) * 16);
        #pragma unroll
        for (int j = 0; j < 8; ++j) {
            v16bf B = bfrag[j * 32];   // fragment ((kc/32)*8+j)*32 + lane
            acc[j] = __builtin_amdgcn_wmma_f32_16x16x32_bf16(false, A, false, B,
                                                             (short)0, acc[j], false, false);
        }
    }
    #pragma unroll
    for (int j = 0; j < 8; ++j) {
        int col = j * 16 + m;
        float bias = bv[col];
        #pragma unroll
        for (int r = 0; r < 8; ++r) {
            int rowi = r + 8 * h;
            VbT[(size_t)col * L + l0 + rowi] = (__bf16)(acc[j][r] + bias);
        }
    }
}

// ---------------------------------------------------------------------------
// Kernel 3: windowed flash attention. One wave per 16-query tile per modality.
// Logits <= 0 always -> no online max; Z = (hi-ce) + sum exp.
// V chunk staged to LDS with gfx1250 async-LDS loads (ASYNCcnt) when available.
// ---------------------------------------------------------------------------
__global__ void attn_kernel(const float* __restrict__ qab, const int* __restrict__ qrg,
                            const float* __restrict__ m1_data, const float* __restrict__ m2_data,
                            const __bf16* __restrict__ VbT1, const __bf16* __restrict__ VbT2,
                            const float* __restrict__ log_tau1, const float* __restrict__ log_tau2,
                            float* __restrict__ out) {
    int mod = blockIdx.y;
    const float* data = mod ? m2_data : m1_data;
    int stride = mod ? 66 : 130;
    const __bf16* VbT = mod ? VbT2 : VbT1;
    float inv_tau = __expf(-(mod ? *log_tau2 : *log_tau1));
    const float* alpha = qab + mod * 2 * T;
    const float* beta  = alpha + T;
    const int* lo_arr = qrg + mod * 3 * T;
    const int* ce_arr = lo_arr + T;
    const int* hi_arr = ce_arr + T;

    int t0 = blockIdx.x * 16;
    int lane = threadIdx.x;
    int m = lane & 15, h = lane >> 4;
    int q = t0 + m;
    float a_q = alpha[q], b_q = beta[q];
    int lo_q = lo_arr[q], ce_q = ce_arr[q], hi_q = hi_arr[q];

    // union window across the 16-query tile (uniform across the wave)
    int lo_t = lo_q, ce_t = ce_q;
    #pragma unroll
    for (int off = 16; off; off >>= 1) {
        lo_t = min(lo_t, __shfl_xor(lo_t, off, 32));
        ce_t = max(ce_t, __shfl_xor(ce_t, off, 32));
    }
    int lstart = lo_t & ~31;

    __shared__ float  dbuf[32];
    __shared__ __bf16 vbufT[128 * 32];   // [col][k] column-major chunk
    __shared__ float  zbuf[16];

    v8f acc[8] = {};
    float psum = 0.f;

    for (int l0 = lstart; l0 < ce_t; l0 += 32) {
        __syncthreads();   // protect LDS reuse across iterations
        // stage key scalar (last feature of mdata)
        dbuf[lane] = data[(size_t)(l0 + lane) * stride + (stride - 1)];
        // stage 32x128 V chunk (column-major): 4 columns x 64B per lane
        int c0 = lane * 4;
#if HAVE_ASYNC_LDS
        #pragma unroll
        for (int cc = 0; cc < 4; ++cc) {
            int4v*   s = (int4v*)(VbT + (size_t)(c0 + cc) * L + l0);   // 64B = 4 x int4
            as3_int4 d = (as3_int4)(as3_void)(vbufT + (c0 + cc) * 32);
            __builtin_amdgcn_global_load_async_to_lds_b128(s,     d,     0, 0);
            __builtin_amdgcn_global_load_async_to_lds_b128(s + 1, d + 1, 0, 0);
            __builtin_amdgcn_global_load_async_to_lds_b128(s + 2, d + 2, 0, 0);
            __builtin_amdgcn_global_load_async_to_lds_b128(s + 3, d + 3, 0, 0);
        }
#else
        #pragma unroll
        for (int cc = 0; cc < 4; ++cc) {
            const uint4* src = (const uint4*)(VbT + (size_t)(c0 + cc) * L + l0);
            uint4* dst = (uint4*)(vbufT + (c0 + cc) * 32);
            dst[0] = src[0]; dst[1] = src[1]; dst[2] = src[2]; dst[3] = src[3];
        }
#endif
        if (l0 + 32 < ce_t)
            __builtin_prefetch(data + (size_t)(l0 + 32 + lane) * stride + (stride - 1), 0, 0);

        // P tile (16x32) in A layout; accumulate row sums for Z
        v16bf A;
        float ps = 0.f;
        __syncthreads();   // dbuf visible (single wave: waits DScnt)
        #pragma unroll
        for (int e = 0; e < 16; ++e) {
            int k = a_kidx(e, h);
            int l = l0 + k;
            float p = 0.f;
            if (l >= lo_q && l < ce_q) {
                float s = a_q - b_q * dbuf[k];
                p = __expf(-(s * s) * inv_tau);
            }
            ps += p;
            A[e] = (__bf16)p;
        }
        psum += ps;
#if HAVE_ASYNC_LDS
#  if HAVE_WAIT_ASYNC
        __builtin_amdgcn_s_wait_asynccnt(0);
#  else
        asm volatile("s_wait_asynccnt 0x0" ::: "memory");
#  endif
#endif
        // O += P @ V : 8 WMMA ops, B fragments are contiguous 32B LDS loads
        #pragma unroll
        for (int j = 0; j < 8; ++j) {
            const v16bf* bp = (const v16bf*)(vbufT + ((j * 16 + m) * 32 + h * 16));
            v16bf B = *bp;
            acc[j] = __builtin_amdgcn_wmma_f32_16x16x32_bf16(false, A, false, B,
                                                             (short)0, acc[j], false, false);
        }
    }

    // denominator: non-causal matching keys contribute exp(0)=1 each
    float rowsum = psum + __shfl_xor(psum, 16, 32);
    if (lane < 16) zbuf[m] = (float)(hi_q - ce_q) + rowsum;
    __syncthreads();

    #pragma unroll
    for (int j = 0; j < 8; ++j) {
        int col = j * 16 + m;
        #pragma unroll
        for (int r = 0; r < 8; ++r) {
            int rowi = r + 8 * h;
            float z = zbuf[rowi];
            out[(size_t)(t0 + rowi) * 256 + mod * 128 + col] = acc[j][r] / z;
        }
    }
}

// ---------------------------------------------------------------------------
extern "C" void kernel_launch(void* const* d_in, const int* in_sizes, int n_in,
                              void* d_out, int out_size, void* d_ws, size_t ws_size,
                              hipStream_t stream) {
    const float* ref_data = (const float*)d_in[0];
    const float* ref_time = (const float*)d_in[1];
    const int*   ref_idx  = (const int*)d_in[2];
    const float* m1_data  = (const float*)d_in[3];
    const float* m1_time  = (const float*)d_in[4];
    const int*   m1_idx   = (const int*)d_in[5];
    const float* m2_data  = (const float*)d_in[6];
    const float* m2_time  = (const float*)d_in[7];
    const int*   m2_idx   = (const int*)d_in[8];
    const float* Wq  = (const float*)d_in[9];
    const float* bq  = (const float*)d_in[10];
    const float* Wk1 = (const float*)d_in[11];
    const float* bk1 = (const float*)d_in[12];
    const float* Wv1 = (const float*)d_in[13];
    const float* bv1 = (const float*)d_in[14];
    const float* Wk2 = (const float*)d_in[15];
    const float* bk2 = (const float*)d_in[16];
    const float* Wv2 = (const float*)d_in[17];
    const float* bv2 = (const float*)d_in[18];
    const float* log_tau1 = (const float*)d_in[19];
    const float* log_tau2 = (const float*)d_in[20];
    float* out = (float*)d_out;

    char* ws = (char*)d_ws;
    float*  P    = (float*)ws;                                   // 2*160 floats
    float*  qab  = (float*)(ws + 2048);                          // 4*T floats
    int*    qrg  = (int*)  (ws + 2048 + 4 * T * 4);              // 6*T ints
    __bf16* VbT1 = (__bf16*)(ws + 2048 + 10 * T * 4);            // 128*L bf16
    __bf16* VbT2 = VbT1 + (size_t)128 * L;                       // 128*L bf16
    __bf16* WvB1 = VbT2 + (size_t)128 * L;                       // 1024 frags * 16
    __bf16* WvB2 = WvB1 + (size_t)1024 * 16;                     // 512 frags * 16

    prep_kernel<<<1, 128, 0, stream>>>(Wq, bq, Wk1, bk1, Wk2, bk2, P);
    wvprep_kernel<<<dim3(4, 2), 256, 0, stream>>>(Wv1, Wv2, WvB1, WvB2);
    qparam_kernel<<<T / 256, 256, 0, stream>>>(ref_data, ref_time, ref_idx,
                                               m1_time, m1_idx, m2_time, m2_idx,
                                               P, qab, qrg);
    vproj_kernel<<<dim3(L / 16, 2), 32, 0, stream>>>(m1_data, WvB1, bv1,
                                                     m2_data, WvB2, bv2, VbT1, VbT2);
    attn_kernel<<<dim3(T / 16, 2), 32, 0, stream>>>(qab, qrg, m1_data, m2_data,
                                                    VbT1, VbT2, log_tau1, log_tau2, out);
}
